// PooledLSTMSummComp_66571993088491
// MI455X (gfx1250) — matrix operand order repaired and checked by reading, not compiled, not run on hardware
//
#include <hip/hip_runtime.h>
#include <hip/hip_bf16.h>
#include <math.h>

// ---------------------------------------------------------------------------
// Bidirectional gated-LSTM + pooling for MI455X (gfx1250, wave32, WMMA).
//   Phase 0: pack f32 -> bf16 (values) and transposed-bf16 weights.
//   Phase 1: xg_f / xg_b / skip GEMMs via v_wmma_f32_16x16x32_bf16.
//   Phase 2: sequential scan, 4 self-contained WGs (dir x 16-batch tile),
//            h in LDS (bf16), c/h state in VGPRs, WhT streamed from L2,
//            per-step xg slab double-buffered into LDS with
//            global_load_async_to_lds_b128 (+ s_wait_asynccnt).
//   Phase 3: masked mean/max pooling.
// ---------------------------------------------------------------------------

typedef __attribute__((ext_vector_type(16))) __bf16 v16bf;
typedef __attribute__((ext_vector_type(8)))  __bf16 v8bf;
typedef __attribute__((ext_vector_type(8)))  float  v8f;

#define B_SZ 32
#define S_SZ 512
#define V_SZ 768
#define H_SZ 512
#define G_SZ 2048                 // 4*H
#define NROW (B_SZ * S_SZ)        // 16384
static constexpr int HSTR = H_SZ + 16;   // LDS h row stride in halves (bank skew)

// ---- WMMA helper ----------------------------------------------------------
__device__ __forceinline__ v8f wmma_bf16(v16bf a, v16bf b, v8f c) {
    return __builtin_amdgcn_wmma_f32_16x16x32_bf16(
        /*neg_a=*/false, a, /*neg_b=*/false, b,
        /*c_mod=*/(short)0, c, /*reuse_a=*/false, /*reuse_b=*/false);
}

// A fragment (16x32 bf16, M-major source, row stride `stride` halves).
// k order used here (contiguous, matches B): lane m=L&15; lanes 0-15 hold K=0..15,
// lanes 16-31 hold K=16..31 (halves 0..15 -> K=kbase+0..15).
__device__ __forceinline__ v16bf load_frag_a(const __bf16* base, int stride,
                                             int lane, int k0) {
    const int m  = lane & 15;
    const int kb = (lane >> 4) << 4;           // 0 or 16
    const __bf16* p = base + (size_t)m * stride + k0 + kb;
    v8bf lo = *(const v8bf*)(p);
    v8bf hi = *(const v8bf*)(p + 8);
    return __builtin_shufflevector(lo, hi, 0,1,2,3,4,5,6,7,
                                           8,9,10,11,12,13,14,15);
}

// B fragment (32x16 bf16) from transposed weights WT[N][K] (row stride K).
// k order used here (contiguous, matches A): lane n=L&15; lanes 0-15 hold K=k0..k0+15, lanes 16-31
// hold K=k0+16..k0+31 -> one contiguous 32B load per lane.
__device__ __forceinline__ v16bf load_frag_b(const __bf16* wT, int strideK,
                                             int n0, int lane, int k0) {
    const int n  = lane & 15;
    const int kb = (lane >> 4) << 4;           // 0 or 16
    return *(const v16bf*)(wT + (size_t)(n0 + n) * strideK + k0 + kb);
}

// ---- CDNA5 async global->LDS copy (ASYNCcnt-tracked) ----------------------
__device__ __forceinline__ void async_load_b128(unsigned lds_off,
                                                const void* gaddr) {
    asm volatile("global_load_async_to_lds_b128 %0, %1, off"
                 :: "v"(lds_off), "v"(gaddr) : "memory");
}
__device__ __forceinline__ void wait_async_le16() {
    asm volatile("s_wait_asynccnt 16" ::: "memory");
}
__device__ __forceinline__ void wait_async_0() {
    asm volatile("s_wait_asynccnt 0" ::: "memory");
}

// ---- Phase 0: converts / transposes ---------------------------------------
__global__ void f32_to_bf16_kernel(const float* __restrict__ in,
                                   __bf16* __restrict__ out, int n) {
    int i = blockIdx.x * blockDim.x + threadIdx.x;
    if (i < n) out[i] = (__bf16)in[i];
}

__global__ void transpose_to_bf16_kernel(const float* __restrict__ W,  // [K][N]
                                         __bf16* __restrict__ WT,      // [N][K]
                                         int K, int N) {
    int i = blockIdx.x * blockDim.x + threadIdx.x;   // dst linear: n*K + k
    if (i >= K * N) return;
    int n = i / K, k = i - n * K;
    WT[i] = (__bf16)W[(size_t)k * N + n];
}

// ---- Phase 1: C[NROW][Ncols] = valbf @ WT^T + bias ------------------------
// One wave per 16x64 output tile block (4 N-tiles share the A fragment).
template <typename OUT_T>
__global__ __launch_bounds__(256) void gemm_xw_kernel(
    const __bf16* __restrict__ Abf,    // [NROW][V_SZ]
    const __bf16* __restrict__ WT,     // [Ncols][V_SZ]
    const float*  __restrict__ bias,   // [Ncols]
    OUT_T* __restrict__ C,             // [NROW][Ncols]
    int Ncols)
{
    const int lane = threadIdx.x & 31;
    const int wv   = threadIdx.x >> 5;
    const int gid  = blockIdx.x * 8 + wv;
    const int n4cnt = Ncols >> 6;
    const int mt = gid / n4cnt;
    const int n4 = gid - mt * n4cnt;
    if (mt >= NROW / 16) return;
    const int row0 = mt * 16, col0 = n4 * 64;
    const int nn = lane & 15, mb = (lane >> 4) * 8;

    v8f acc[4];
    #pragma unroll
    for (int j = 0; j < 4; ++j) {
        float bj = bias[col0 + j * 16 + nn];
        #pragma unroll
        for (int r = 0; r < 8; ++r) acc[j][r] = bj;
    }
    for (int kt = 0; kt < V_SZ / 32; ++kt) {
        v16bf a = load_frag_a(Abf + (size_t)row0 * V_SZ, V_SZ, lane, kt * 32);
        #pragma unroll
        for (int j = 0; j < 4; ++j) {
            v16bf b = load_frag_b(WT, V_SZ, col0 + j * 16, lane, kt * 32);
            acc[j] = wmma_bf16(a, b, acc[j]);
        }
    }
    #pragma unroll
    for (int j = 0; j < 4; ++j)
        #pragma unroll
        for (int r = 0; r < 8; ++r)
            C[(size_t)(row0 + mb + r) * Ncols + col0 + j * 16 + nn] =
                (OUT_T)acc[j][r];
}

// ---- Phase 2: gated LSTM scan ---------------------------------------------
// grid = 4 blocks: blockIdx = dir*2 + batch_tile. 8 waves; wave w owns
// H-columns [64w, 64w+64) => gate columns {g*512 + 64w + ...} so i/f/g/o for
// an (m,n) element land in the same lane/VGPR slot as its private c/h state.
// Per-step xg slab (16 rows x 2048 gates, bf16 = 64KB) is double-buffered in
// LDS via async global->LDS b128 copies issued one step ahead.
__global__ __launch_bounds__(256) void lstm_scan_kernel(
    const __bf16* __restrict__ xg_f, const __bf16* __restrict__ xg_b,
    const __bf16* __restrict__ WhT_f, const __bf16* __restrict__ WhT_b,
    const float* __restrict__ alphas,     // [B][S]
    float* __restrict__ out_f, float* __restrict__ out_b)
{
    __shared__ __bf16 hbuf[16 * HSTR];           // 16.5 KB
    __shared__ __bf16 xgls[2 * 16 * G_SZ];       // 128 KB double buffer

    const int lane = threadIdx.x & 31;
    const int wv   = threadIdx.x >> 5;
    const int dir  = blockIdx.x >> 1;
    const int b0   = (blockIdx.x & 1) * 16;
    const __bf16* xg  = dir ? xg_b : xg_f;
    const __bf16* WhT = dir ? WhT_b : WhT_f;
    float* outp = dir ? out_b : out_f;

    const int nh0 = wv * 64;
    const int nn  = lane & 15;
    const int mb  = (lane >> 4) * 8;
    const unsigned xgls_off = (unsigned)(size_t)(&xgls[0]);

    for (int i = threadIdx.x; i < 16 * HSTR; i += 256) hbuf[i] = (__bf16)0.0f;

    float c_reg[4][8], h_reg[4][8];
    #pragma unroll
    for (int q = 0; q < 4; ++q)
        #pragma unroll
        for (int r = 0; r < 8; ++r) { c_reg[q][r] = 0.f; h_reg[q][r] = 0.f; }

    // Prologue: stage step 0's xg slab into buffer 0 (16 x b128 per thread).
    {
        const int s0 = dir ? (S_SZ - 1) : 0;
        #pragma unroll
        for (int j = 0; j < 16; ++j) {
            int e   = (threadIdx.x + j * 256) * 8;   // halves
            int row = e >> 11, col = e & 2047;
            const __bf16* gp =
                xg + ((size_t)(b0 + row) * S_SZ + s0) * (size_t)G_SZ + col;
            async_load_b128(xgls_off + (unsigned)(e * 2), gp);
        }
    }

    for (int t = 0; t < S_SZ; ++t) {
        const int s = dir ? (S_SZ - 1 - t) : t;

        // (1) Kick off async copy of next step's xg slab into the other buffer.
        if (t + 1 < S_SZ) {
            const int s2 = dir ? (S_SZ - 2 - t) : (t + 1);
            const unsigned bufo = ((t + 1) & 1) ? (unsigned)(16 * G_SZ * 2) : 0u;
            #pragma unroll
            for (int j = 0; j < 16; ++j) {
                int e   = (threadIdx.x + j * 256) * 8;
                int row = e >> 11, col = e & 2047;
                const __bf16* gp =
                    xg + ((size_t)(b0 + row) * S_SZ + s2) * (size_t)G_SZ + col;
                async_load_b128(xgls_off + bufo + (unsigned)(e * 2), gp);
            }
            wait_async_le16();   // in-order: only the 16 just-issued remain
        } else {
            wait_async_0();
        }
        __syncthreads();         // current slab + prev-step h visible to all

        const __bf16* xcur = &xgls[(t & 1) * (16 * G_SZ)];

        float av[8];
        #pragma unroll
        for (int r = 0; r < 8; ++r)
            av[r] = alphas[(size_t)(b0 + mb + r) * S_SZ + s];

        #pragma unroll
        for (int q = 0; q < 4; ++q) {
            v8f acc[4];
            #pragma unroll
            for (int r = 0; r < 8; ++r) {
                const __bf16* xp = xcur + (mb + r) * G_SZ + nh0 + q * 16 + nn;
                acc[0][r] = (float)xp[0];            // i
                acc[1][r] = (float)xp[512];          // f
                acc[2][r] = (float)xp[1024];         // g
                acc[3][r] = (float)xp[1536];         // o
            }
            #pragma unroll
            for (int kh = 0; kh < 2; ++kh) {         // two K halves of 8 tiles
                v16bf a8[8];
                #pragma unroll
                for (int k8 = 0; k8 < 8; ++k8)
                    a8[k8] = load_frag_a(hbuf, HSTR, lane, (kh * 8 + k8) * 32);
                #pragma unroll
                for (int k8 = 0; k8 < 8; ++k8) {
                    const int k0 = (kh * 8 + k8) * 32;
                    #pragma unroll
                    for (int g = 0; g < 4; ++g) {
                        v16bf b = load_frag_b(WhT, H_SZ,
                                              g * 512 + nh0 + q * 16, lane, k0);
                        acc[g] = wmma_bf16(a8[k8], b, acc[g]);
                    }
                }
            }
            #pragma unroll
            for (int r = 0; r < 8; ++r) {
                float iv = acc[0][r], fv = acc[1][r];
                float gv = acc[2][r], ov = acc[3][r];
                float co = c_reg[q][r], ho = h_reg[q][r];
                float si = 1.f / (1.f + __expf(-iv));
                float sf = 1.f / (1.f + __expf(-fv));
                float so = 1.f / (1.f + __expf(-ov));
                float cn = sf * co + si * tanhf(gv);
                float hn = so * tanhf(cn);
                float a_ = av[r];
                float c2 = a_ * cn + (1.f - a_) * co;
                float h2 = a_ * hn + (1.f - a_) * ho;
                c_reg[q][r] = c2;
                h_reg[q][r] = h2;
                outp[((size_t)(b0 + mb + r) * S_SZ + s) * (size_t)H_SZ
                     + nh0 + q * 16 + nn] = h2;
            }
        }

        __syncthreads();         // all hbuf reads of old h are done
        // (2) Publish new h (own columns only) for next step's A fragments.
        #pragma unroll
        for (int q = 0; q < 4; ++q)
            #pragma unroll
            for (int r = 0; r < 8; ++r)
                hbuf[(mb + r) * HSTR + nh0 + q * 16 + nn] =
                    (__bf16)h_reg[q][r];
        // visibility of these writes is covered by next step's top barrier
    }
}

// ---- Phase 3: masked mean/max pooling -------------------------------------
__global__ void pool_kernel(const float* __restrict__ out_f,
                            const float* __restrict__ out_b,
                            const float* __restrict__ xs,      // [NROW][2H]
                            const float* __restrict__ alphas,  // [B][S]
                            float* __restrict__ out)           // [B][4H]
{
    int idx = blockIdx.x * blockDim.x + threadIdx.x;   // b*1024 + j
    if (idx >= B_SZ * 2 * H_SZ) return;
    int b = idx >> 10, j = idx & 1023;
    const float* top = (j < H_SZ) ? out_f : out_b;
    int jj = (j < H_SZ) ? j : (j - H_SZ);
    float sum = 0.f, asum = 0.f, mx = -INFINITY;
    for (int s = 0; s < S_SZ; ++s) {
        float a = alphas[(size_t)b * S_SZ + s];
        float r = (top[((size_t)b * S_SZ + s) * H_SZ + jj] +
                   xs[((size_t)b * S_SZ + s) * (2 * H_SZ) + j]) * a;
        sum += r; asum += a;
        if (a > 0.f) mx = fmaxf(mx, r);
    }
    if (mx == -INFINITY) mx = 0.f;
    out[(size_t)b * (4 * H_SZ) + j] = sum / (asum + 1e-6f);
    out[(size_t)b * (4 * H_SZ) + 2 * H_SZ + j] = mx;
}

// ---------------------------------------------------------------------------
extern "C" void kernel_launch(void* const* d_in, const int* in_sizes, int n_in,
                              void* d_out, int out_size, void* d_ws, size_t ws_size,
                              hipStream_t stream)
{
    const float* values = (const float*)d_in[0];   // [B,S,V]
    const float* alphas = (const float*)d_in[1];   // [B,S]
    const float* Wx_f   = (const float*)d_in[2];   // [V,4H]
    const float* Wh_f   = (const float*)d_in[3];   // [H,4H]
    const float* b_f    = (const float*)d_in[4];   // [4H]
    const float* Wx_b   = (const float*)d_in[5];
    const float* Wh_b   = (const float*)d_in[6];
    const float* b_b    = (const float*)d_in[7];
    const float* Ws     = (const float*)d_in[8];   // [V,2H]
    const float* bs     = (const float*)d_in[9];   // [2H]
    float* out = (float*)d_out;

    char* ws = (char*)d_ws;
    size_t off = 0;
    auto alloc = [&](size_t bytes) {
        void* p = ws + off;
        off = (off + bytes + 255) & ~(size_t)255;
        return p;
    };
    __bf16* valbf = (__bf16*)alloc((size_t)NROW * V_SZ * 2);      // 25.2 MB
    __bf16* WxTf  = (__bf16*)alloc((size_t)G_SZ * V_SZ * 2);      //  3.1 MB
    __bf16* WxTb  = (__bf16*)alloc((size_t)G_SZ * V_SZ * 2);
    __bf16* WsT   = (__bf16*)alloc((size_t)(2 * H_SZ) * V_SZ * 2);
    __bf16* WhTf  = (__bf16*)alloc((size_t)G_SZ * H_SZ * 2);      //  2.1 MB
    __bf16* WhTb  = (__bf16*)alloc((size_t)G_SZ * H_SZ * 2);
    __bf16* xg_f  = (__bf16*)alloc((size_t)NROW * G_SZ * 2);      // 67.1 MB
    __bf16* xg_b  = (__bf16*)alloc((size_t)NROW * G_SZ * 2);
    float*  xs    = (float*) alloc((size_t)NROW * 2 * H_SZ * 4);  // 67.1 MB
    float*  out_f = (float*) alloc((size_t)NROW * H_SZ * 4);      // 33.6 MB
    float*  out_b = (float*) alloc((size_t)NROW * H_SZ * 4);
    (void)ws_size; (void)in_sizes; (void)n_in; (void)out_size;

    // Phase 0: converts / transposes
    {
        int n = NROW * V_SZ;
        f32_to_bf16_kernel<<<(n + 255) / 256, 256, 0, stream>>>(values, valbf, n);
    }
    transpose_to_bf16_kernel<<<(V_SZ * G_SZ + 255) / 256, 256, 0, stream>>>(
        Wx_f, WxTf, V_SZ, G_SZ);
    transpose_to_bf16_kernel<<<(V_SZ * G_SZ + 255) / 256, 256, 0, stream>>>(
        Wx_b, WxTb, V_SZ, G_SZ);
    transpose_to_bf16_kernel<<<(V_SZ * 2 * H_SZ + 255) / 256, 256, 0, stream>>>(
        Ws, WsT, V_SZ, 2 * H_SZ);
    transpose_to_bf16_kernel<<<(H_SZ * G_SZ + 255) / 256, 256, 0, stream>>>(
        Wh_f, WhTf, H_SZ, G_SZ);
    transpose_to_bf16_kernel<<<(H_SZ * G_SZ + 255) / 256, 256, 0, stream>>>(
        Wh_b, WhTb, H_SZ, G_SZ);

    // Phase 1: xg GEMMs (bf16 out) + skip GEMM (f32 out)
    {
        int blocks_xg = (NROW / 16) * (G_SZ / 64) / 8;        // 4096
        gemm_xw_kernel<__bf16><<<blocks_xg, 256, 0, stream>>>(
            valbf, WxTf, b_f, xg_f, G_SZ);
        gemm_xw_kernel<__bf16><<<blocks_xg, 256, 0, stream>>>(
            valbf, WxTb, b_b, xg_b, G_SZ);
        int blocks_xs = (NROW / 16) * ((2 * H_SZ) / 64) / 8;  // 2048
        gemm_xw_kernel<float><<<blocks_xs, 256, 0, stream>>>(
            valbf, WsT, bs, xs, 2 * H_SZ);
    }

    // Phase 2: the sequential scan (4 self-contained workgroups)
    lstm_scan_kernel<<<4, 256, 0, stream>>>(xg_f, xg_b, WhTf, WhTb, alphas,
                                            out_f, out_b);

    // Phase 3: pooling
    pool_kernel<<<(B_SZ * 2 * H_SZ + 255) / 256, 256, 0, stream>>>(
        out_f, out_b, xs, alphas, out);
}
